// MHNN_61297773248566
// MI455X (gfx1250) — compile-verified, matrix operand order checked by reading, not run on hardware
//
#include <hip/hip_runtime.h>
#include <hip/hip_bf16.h>

typedef __bf16 bf16;
typedef __attribute__((ext_vector_type(16))) __bf16 v16bf;
typedef __attribute__((ext_vector_type(8)))  __bf16 v8bf;
typedef __attribute__((ext_vector_type(8)))  float  v8f;

#define T_STEPS 24
#define BATCH   64
#define DIM     1536
#define RNEUR   5120
#define CLASSES 100
#define NBLK    5
#define NBN     1024
#define KCHUNKS 8
#define KCH     (RNEUR / KCHUNKS)   // 640

__device__ __forceinline__ float clamp01(float x) { return fminf(fmaxf(x, 0.f), 1.f); }

// ---------------------------------------------------------------------------
// LayerNorm over D, emit bf16 normalized activations
// ---------------------------------------------------------------------------
__global__ void layernorm_bf16_kernel(const float* __restrict__ x,
                                      const float* __restrict__ g,
                                      const float* __restrict__ bsh,
                                      bf16* __restrict__ xn) {
    __shared__ float s1[256], s2[256];
    const int row = blockIdx.x;  // t*B + b
    const float* xr = x + (size_t)row * DIM;
    float sum = 0.f, sq = 0.f;
    for (int i = threadIdx.x; i < DIM; i += 256) { float v = xr[i]; sum += v; sq += v * v; }
    s1[threadIdx.x] = sum; s2[threadIdx.x] = sq;
    __syncthreads();
    for (int off = 128; off > 0; off >>= 1) {
        if ((int)threadIdx.x < off) { s1[threadIdx.x] += s1[threadIdx.x + off]; s2[threadIdx.x] += s2[threadIdx.x + off]; }
        __syncthreads();
    }
    const float mean = s1[0] / (float)DIM;
    const float var  = s2[0] / (float)DIM - mean * mean;
    const float inv  = rsqrtf(var + 1e-5f);
    bf16* xo = xn + (size_t)row * DIM;
    for (int i = threadIdx.x; i < DIM; i += 256) {
        float v = (xr[i] - mean) * inv * g[i] + bsh[i];
        xo[i] = (bf16)v;
    }
}

// ---------------------------------------------------------------------------
// Masked weight -> bf16
// ---------------------------------------------------------------------------
__global__ void maskmul_bf16_kernel(const float* __restrict__ w, const float* __restrict__ m,
                                    bf16* __restrict__ o, long n) {
    long i = (long)blockIdx.x * blockDim.x + threadIdx.x;
    if (i < n) o[i] = (bf16)(w[i] * m[i]);
}

// ---------------------------------------------------------------------------
// Zero scan state
// ---------------------------------------------------------------------------
__global__ void init_state_kernel(float* mem, float* spk, bf16* spkb,
                                  float* ttr, float* rtr, float* ctr, float* ssum) {
    int i = blockIdx.x * blockDim.x + threadIdx.x;
    mem[i] = 0.f; spk[i] = 0.f; spkb[i] = (bf16)0.f;
    ttr[i] = 0.f; rtr[i] = 0.f; ctr[i] = 0.f; ssum[i] = 0.f;
}

// ---------------------------------------------------------------------------
// CDNA5 16-bit fragment load: lane L holds row (L&15), K = (L>>4)*8 + {0..7, 16..23}
// => two contiguous b128 loads per lane.
// ---------------------------------------------------------------------------
__device__ __forceinline__ v16bf load_frag16(const bf16* p) {
    v8bf lo = *(const v8bf*)(p);
    v8bf hi = *(const v8bf*)(p + 16);
    v16bf r;
#pragma unroll
    for (int i = 0; i < 8; ++i) { r[i] = lo[i]; r[i + 8] = hi[i]; }
    return r;
}

#define WMMA_BF16(A_, B_, C_) \
    __builtin_amdgcn_wmma_f32_16x16x32_bf16(false, (A_), false, (B_), (short)0, (C_), false, false)

// ---------------------------------------------------------------------------
// Ping-pong software-pipelined main loop: one A fragment reused against 4 B
// fragments per k-step; loads for the next step go into the *other* buffer
// set, so there are no register-rotation copies (round 4 showed the copy
// variant costs ~40 v_mov per iteration). "#pragma unroll 1" prevents the
// unroller from multiplying live buffers into scratch spills (round 3).
// ---------------------------------------------------------------------------
template <bool PF>
__device__ __forceinline__ void gemm_mainloop(const bf16* __restrict__ aPtr,
                                              const bf16* __restrict__ bPtr,
                                              size_t ldb, int klen, v8f* acc) {
    const size_t s1 = (size_t)16 * ldb;
    const size_t s2 = 2 * s1, s3 = 3 * s1;
    v16bf a0 = load_frag16(aPtr);
    v16bf p0 = load_frag16(bPtr);
    v16bf p1 = load_frag16(bPtr + s1);
    v16bf p2 = load_frag16(bPtr + s2);
    v16bf p3 = load_frag16(bPtr + s3);
    int kb = 32;
#pragma unroll 1
    for (; kb + 32 < klen; kb += 64) {
        v16bf a1 = load_frag16(aPtr + kb);
        v16bf q0 = load_frag16(bPtr + kb);
        v16bf q1 = load_frag16(bPtr + s1 + kb);
        v16bf q2 = load_frag16(bPtr + s2 + kb);
        v16bf q3 = load_frag16(bPtr + s3 + kb);
        if (PF) {
            __builtin_prefetch(aPtr + kb + 1024, 0, 3);
            __builtin_prefetch(bPtr + kb + 1024, 0, 3);
            __builtin_prefetch(bPtr + s1 + kb + 1024, 0, 3);
            __builtin_prefetch(bPtr + s2 + kb + 1024, 0, 3);
            __builtin_prefetch(bPtr + s3 + kb + 1024, 0, 3);
        }
        acc[0] = WMMA_BF16(a0, p0, acc[0]);
        acc[1] = WMMA_BF16(a0, p1, acc[1]);
        acc[2] = WMMA_BF16(a0, p2, acc[2]);
        acc[3] = WMMA_BF16(a0, p3, acc[3]);
        a0 = load_frag16(aPtr + kb + 32);
        p0 = load_frag16(bPtr + kb + 32);
        p1 = load_frag16(bPtr + s1 + kb + 32);
        p2 = load_frag16(bPtr + s2 + kb + 32);
        p3 = load_frag16(bPtr + s3 + kb + 32);
        acc[0] = WMMA_BF16(a1, q0, acc[0]);
        acc[1] = WMMA_BF16(a1, q1, acc[1]);
        acc[2] = WMMA_BF16(a1, q2, acc[2]);
        acc[3] = WMMA_BF16(a1, q3, acc[3]);
    }
    if (kb < klen) {
        // one leftover step: load it, drain buffer 0, then buffer 1
        v16bf a1 = load_frag16(aPtr + kb);
        v16bf q0 = load_frag16(bPtr + kb);
        v16bf q1 = load_frag16(bPtr + s1 + kb);
        v16bf q2 = load_frag16(bPtr + s2 + kb);
        v16bf q3 = load_frag16(bPtr + s3 + kb);
        acc[0] = WMMA_BF16(a0, p0, acc[0]);
        acc[1] = WMMA_BF16(a0, p1, acc[1]);
        acc[2] = WMMA_BF16(a0, p2, acc[2]);
        acc[3] = WMMA_BF16(a0, p3, acc[3]);
        acc[0] = WMMA_BF16(a1, q0, acc[0]);
        acc[1] = WMMA_BF16(a1, q1, acc[1]);
        acc[2] = WMMA_BF16(a1, q2, acc[2]);
        acc[3] = WMMA_BF16(a1, q3, acc[3]);
    } else {
        acc[0] = WMMA_BF16(a0, p0, acc[0]);
        acc[1] = WMMA_BF16(a0, p1, acc[1]);
        acc[2] = WMMA_BF16(a0, p2, acc[2]);
        acc[3] = WMMA_BF16(a0, p3, acc[3]);
    }
}

// ---------------------------------------------------------------------------
// drive = xn @ WpMasked^T + bp ; each wave owns a 16(M) x 64(N) strip.
// grid (M/32, N/256), block 256 (8 waves: 2 M-tiles x 4 N-strips)
// ---------------------------------------------------------------------------
__global__ void __attribute__((amdgpu_waves_per_eu(1)))
gemm_drive_wmma(const bf16* __restrict__ A,
                const bf16* __restrict__ Bw,
                const float* __restrict__ bp,
                float* __restrict__ drive) {
    const int lane = threadIdx.x & 31;
    const int wave = threadIdx.x >> 5;
    const int mt = blockIdx.x * 2 + (wave & 1);
    const int ns = blockIdx.y * 4 + (wave >> 1);
    const int m0 = mt * 16;
    const int n0 = ns * 64;
    const int lrow = lane & 15;
    const int koff = (lane >> 4) * 8;
    const bf16* aPtr = A  + (size_t)(m0 + lrow) * DIM + koff;
    const bf16* bPtr = Bw + (size_t)(n0 + lrow) * DIM + koff;
    v8f acc[4] = {{}, {}, {}, {}};
    gemm_mainloop<true>(aPtr, bPtr, DIM, DIM, acc);
    const int mrow = m0 + ((lane >> 4) * 8);
#pragma unroll
    for (int q = 0; q < 4; ++q) {
        const int n = n0 + q * 16 + lrow;
        const float bias = bp[n];
#pragma unroll
        for (int v = 0; v < 8; ++v)
            drive[(size_t)(mrow + v) * RNEUR + n] = acc[q][v] + bias;
    }
}

// ---------------------------------------------------------------------------
// lateral partials: lat_part[kc] = spike @ WlMasked^T over K-chunk kc.
// grid (2, 20, KCHUNKS), block 256 -> 2560 waves/step. Deterministic split-K.
// Wl (52 MB bf16) is L2-resident on MI455X (192 MB L2) -> no prefetch.
// ---------------------------------------------------------------------------
__global__ void __attribute__((amdgpu_waves_per_eu(1)))
gemm_lateral_wmma(const bf16* __restrict__ Sb,
                  const bf16* __restrict__ Wl,
                  float* __restrict__ lat_part) {
    const int lane = threadIdx.x & 31;
    const int wave = threadIdx.x >> 5;
    const int mt = blockIdx.x * 2 + (wave & 1);   // 0..3 (batch tiles)
    const int ns = blockIdx.y * 4 + (wave >> 1);  // 0..79 (neuron strips)
    const int kc = blockIdx.z;
    const int m0 = mt * 16;
    const int n0 = ns * 64;
    const int lrow = lane & 15;
    const int koff = (lane >> 4) * 8;
    const bf16* aPtr = Sb + (size_t)(m0 + lrow) * RNEUR + kc * KCH + koff;
    const bf16* bPtr = Wl + (size_t)(n0 + lrow) * RNEUR + kc * KCH + koff;
    v8f acc[4] = {{}, {}, {}, {}};
    gemm_mainloop<false>(aPtr, bPtr, RNEUR, KCH, acc);
    float* out = lat_part + (size_t)kc * BATCH * RNEUR;
    const int b0 = m0 + ((lane >> 4) * 8);
#pragma unroll
    for (int q = 0; q < 4; ++q) {
        const int n = n0 + q * 16 + lrow;
#pragma unroll
        for (int v = 0; v < 8; ++v)
            out[(size_t)(b0 + v) * RNEUR + n] = acc[q][v];
    }
}

// ---------------------------------------------------------------------------
// Fused LIF + WTA quantile + spike/trace updates.
// Phase 1: mem_pre = (mem - spike*(0.5+ref_beta*rtr))*clip(decay)
//                    + (drive_t + sum_kc lat_part[kc] + bl) + cur_beta*ctr
// Phase 2: exact 0.8-quantile via LDS bitonic sort (jnp linear interp at
//          sorted indices 818/819), ReLU subtract, spike, trace updates.
// One workgroup per (block k, batch b); group elements r = n*5 + k (disjoint).
// ---------------------------------------------------------------------------
__global__ void wta_spike_kernel(const float* __restrict__ lat_part,
                                 const float* __restrict__ drive_t,
                                 const float* __restrict__ bl,
                                 const float* __restrict__ decay,
                                 const float* __restrict__ ref_beta,
                                 const float* __restrict__ cur_beta,
                                 const float* __restrict__ thr_beta,
                                 const float* __restrict__ thr_decay,
                                 const float* __restrict__ ref_decay,
                                 const float* __restrict__ cur_decay,
                                 float* __restrict__ mem,
                                 float* __restrict__ spk,
                                 bf16* __restrict__ spkb,
                                 float* __restrict__ ttr,
                                 float* __restrict__ rtr,
                                 float* __restrict__ ctr,
                                 float* __restrict__ ssum) {
    __shared__ float sh[NBN];    // sorted copy
    __shared__ float shv[NBN];   // original mem_pre values
    __shared__ float s_nps;
    const int blk = blockIdx.x;  // 0..4
    const int b   = blockIdx.y;  // 0..63
    // Phase 1: LIF membrane update (reads carry state, writes LDS only)
    for (int n = threadIdx.x; n < NBN; n += blockDim.x) {
        const int r = n * NBLK + blk;
        const size_t idx = (size_t)b * RNEUR + r;
        float lat = 0.f;
#pragma unroll
        for (int q = 0; q < KCHUNKS; ++q)
            lat += lat_part[(size_t)q * BATCH * RNEUR + idx];
        float st  = drive_t[idx] + lat + bl[r];
        float ref = 0.5f + ref_beta[r] * rtr[idx];
        float cur = cur_beta[r] * ctr[idx];
        float mp  = (mem[idx] - spk[idx] * ref) * clamp01(decay[r]) + st + cur;
        sh[n] = mp; shv[n] = mp;
    }
    __syncthreads();
    // Phase 2a: bitonic sort ascending
    for (unsigned k = 2; k <= NBN; k <<= 1) {
        for (unsigned j = k >> 1; j > 0; j >>= 1) {
            for (unsigned t = threadIdx.x; t < NBN; t += blockDim.x) {
                unsigned ixj = t ^ j;
                if (ixj > t) {
                    float a = sh[t], c = sh[ixj];
                    bool up = ((t & k) == 0);
                    if (up ? (a > c) : (a < c)) { sh[t] = c; sh[ixj] = a; }
                }
            }
            __syncthreads();
        }
    }
    if (threadIdx.x == 0) s_nps = sh[818] + 0.4f * (sh[819] - sh[818]);
    __syncthreads();
    const float nps = s_nps;
    // Phase 2b: WTA subtract, spike, carry-state updates
    for (int n = threadIdx.x; n < NBN; n += blockDim.x) {
        const int r = n * NBLK + blk;
        const size_t idx = (size_t)b * RNEUR + r;
        float m = fmaxf(shv[n] - nps, 0.f);
        float thr = 0.5f + thr_beta[r] * ttr[idx];
        float s = (m - thr > 0.f) ? 1.f : 0.f;
        mem[idx] = m;
        ttr[idx] = ttr[idx] * clamp01(thr_decay[r]) + s;
        rtr[idx] = rtr[idx] * clamp01(ref_decay[r]) + s;
        ctr[idx] = ctr[idx] * clamp01(cur_decay[r]) + s;
        ssum[idx] += s;
        spk[idx]  = s;
        spkb[idx] = (bf16)s;
    }
}

// ---------------------------------------------------------------------------
// Readout: out[b,c] = (ssum[b,:]/T) . Wm[c,:] + bm[c]   (tiny; Wm L2-resident)
// ---------------------------------------------------------------------------
__global__ void readout_kernel(const float* __restrict__ ssum,
                               const float* __restrict__ Wm,
                               const float* __restrict__ bm,
                               float* __restrict__ out) {
    const int b = blockIdx.x;
    const int c = threadIdx.x;
    if (c >= CLASSES) return;
    const float* s = ssum + (size_t)b * RNEUR;
    const float* w = Wm + (size_t)c * RNEUR;
    float acc = 0.f;
    for (int r = 0; r < RNEUR; ++r) acc += s[r] * w[r];
    out[b * CLASSES + c] = acc * (1.f / (float)T_STEPS) + bm[c];
}

// ---------------------------------------------------------------------------
extern "C" void kernel_launch(void* const* d_in, const int* in_sizes, int n_in,
                              void* d_out, int out_size, void* d_ws, size_t ws_size,
                              hipStream_t stream) {
    const float* x        = (const float*)d_in[0];
    const float* ln_g     = (const float*)d_in[1];
    const float* ln_b     = (const float*)d_in[2];
    const float* Wp       = (const float*)d_in[3];
    const float* bp       = (const float*)d_in[4];
    const float* Wl       = (const float*)d_in[5];
    const float* bl       = (const float*)d_in[6];
    const float* Wm       = (const float*)d_in[7];
    const float* bm       = (const float*)d_in[8];
    const float* decay    = (const float*)d_in[9];
    const float* thr_beta = (const float*)d_in[10];
    const float* thr_dec  = (const float*)d_in[11];
    const float* ref_beta = (const float*)d_in[12];
    const float* ref_dec  = (const float*)d_in[13];
    const float* cur_beta = (const float*)d_in[14];
    const float* cur_dec  = (const float*)d_in[15];
    const float* pmask    = (const float*)d_in[16];
    const float* lmask    = (const float*)d_in[17];

    char* ws = (char*)d_ws;
    size_t off = 0;
    auto alloc = [&](size_t bytes) -> void* {
        void* p = ws + off;
        off = (off + bytes + 255) & ~(size_t)255;
        return p;
    };

    bf16*  xn   = (bf16*) alloc((size_t)T_STEPS * BATCH * DIM * sizeof(bf16));
    bf16*  WpB  = (bf16*) alloc((size_t)RNEUR * DIM * sizeof(bf16));
    bf16*  WlB  = (bf16*) alloc((size_t)RNEUR * RNEUR * sizeof(bf16));
    float* drv  = (float*)alloc((size_t)T_STEPS * BATCH * RNEUR * sizeof(float));
    float* latp = (float*)alloc((size_t)KCHUNKS * BATCH * RNEUR * sizeof(float));
    float* mem  = (float*)alloc((size_t)BATCH * RNEUR * sizeof(float));
    float* spk  = (float*)alloc((size_t)BATCH * RNEUR * sizeof(float));
    bf16*  spkb = (bf16*) alloc((size_t)BATCH * RNEUR * sizeof(bf16));
    float* ttr  = (float*)alloc((size_t)BATCH * RNEUR * sizeof(float));
    float* rtr  = (float*)alloc((size_t)BATCH * RNEUR * sizeof(float));
    float* ctr  = (float*)alloc((size_t)BATCH * RNEUR * sizeof(float));
    float* ssum = (float*)alloc((size_t)BATCH * RNEUR * sizeof(float));

    // 1. LayerNorm + bf16 cast
    layernorm_bf16_kernel<<<T_STEPS * BATCH, 256, 0, stream>>>(x, ln_g, ln_b, xn);

    // 2. Masked weights -> bf16
    {
        long nPD = (long)RNEUR * DIM;
        maskmul_bf16_kernel<<<(unsigned)((nPD + 255) / 256), 256, 0, stream>>>(Wp, pmask, WpB, nPD);
        long nLL = (long)RNEUR * RNEUR;
        maskmul_bf16_kernel<<<(unsigned)((nLL + 255) / 256), 256, 0, stream>>>(Wl, lmask, WlB, nLL);
    }

    // 3. Zero scan state
    init_state_kernel<<<(BATCH * RNEUR) / 256, 256, 0, stream>>>(mem, spk, spkb, ttr, rtr, ctr, ssum);

    // 4. Feedforward drive for all T steps (WMMA, 16x64 strips, prefetch)
    gemm_drive_wmma<<<dim3((T_STEPS * BATCH) / 32, RNEUR / 256), 256, 0, stream>>>(xn, WpB, bp, drv);

    // 5. Sequential scan: split-K lateral WMMA GEMM, then fused LIF+WTA+spike
    for (int t = 0; t < T_STEPS; ++t) {
        gemm_lateral_wmma<<<dim3(BATCH / 32, RNEUR / 256, KCHUNKS), 256, 0, stream>>>(spkb, WlB, latp);
        wta_spike_kernel<<<dim3(NBLK, BATCH), 512, 0, stream>>>(
            latp, drv + (size_t)t * BATCH * RNEUR, bl,
            decay, ref_beta, cur_beta, thr_beta, thr_dec, ref_dec, cur_dec,
            mem, spk, spkb, ttr, rtr, ctr, ssum);
    }

    // 6. Readout
    readout_kernel<<<BATCH, 128, 0, stream>>>(ssum, Wm, bm, (float*)d_out);
}
